// MultiHeadedAttention_6734508720603
// MI455X (gfx1250) — compile-verified, matrix operand order checked by reading, not run on hardware
//
#include <hip/hip_runtime.h>
#include <stdint.h>

// ---------------------------------------------------------------------------
// Problem constants (match reference)
// ---------------------------------------------------------------------------
#define Bsz   8
#define Ssz   512
#define Dsz   1024
#define Hsz   16
#define DPH   64
#define MM    (Bsz * Ssz)      // 4096 rows for all big GEMMs
#define CCdim (5 * Dsz)        // 5120 concat width
#define BHS   (Bsz * Hsz * Ssz)// 65536

typedef __bf16 bf16;
typedef __attribute__((ext_vector_type(16))) __bf16 v16bf;
typedef __attribute__((ext_vector_type(8)))  float  v8f;

typedef union { uint4 u4; bf16 h[8]; } B8;          // 16B = 8 bf16
typedef union { struct { B8 lo; B8 hi; } p; v16bf v; } F16; // full fragment
typedef union { float4 v4; float f[4]; } F4;

__device__ __forceinline__ float bf2f(bf16 x) {
  unsigned short u; __builtin_memcpy(&u, &x, 2);
  unsigned int v = ((unsigned int)u) << 16;
  float f; __builtin_memcpy(&f, &v, 4); return f;
}
__device__ __forceinline__ bf16 f2bf(float f) {
  unsigned int u; __builtin_memcpy(&u, &f, 4);
  u += 0x7fffu + ((u >> 16) & 1u);             // round-to-nearest-even
  unsigned short s = (unsigned short)(u >> 16);
  bf16 b; __builtin_memcpy(&b, &s, 2); return b;
}
__device__ __forceinline__ v8f wmma_bf16(v16bf a, v16bf b, v8f c) {
  // D(16x16 f32) = A(16x32 bf16) * B(32x16 bf16) + C
  return __builtin_amdgcn_wmma_f32_16x16x32_bf16(false, a, false, b,
                                                 (short)0, c, false, false);
}

// ---------------------------------------------------------------------------
// f32 -> bf16 conversion (grid-stride)
// ---------------------------------------------------------------------------
__global__ void cvt_f32_bf16(const float* __restrict__ s, bf16* __restrict__ d, int n) {
  for (int i = blockIdx.x * blockDim.x + threadIdx.x; i < n; i += gridDim.x * blockDim.x)
    d[i] = f2bf(s[i]);
}

// ---------------------------------------------------------------------------
// v (b,h,s,dph) -> vT (b,h,dph,s) so attn@v B-fragments read contiguously
// ---------------------------------------------------------------------------
__global__ __launch_bounds__(256) void transpose_v(const bf16* __restrict__ src,
                                                   bf16* __restrict__ dst) {
  const int bh = blockIdx.x;                 // 0 .. B*H-1
  const bf16* s = src + (size_t)bh * Ssz * DPH;
  bf16* d = dst + (size_t)bh * DPH * Ssz;
  for (int i = threadIdx.x; i < Ssz * DPH; i += 256) {
    const int dd = i >> 9, ss = i & 511;     // write-coalesced over s
    d[dd * Ssz + ss] = s[ss * DPH + dd];
  }
}

// ---------------------------------------------------------------------------
// Generic bf16 WMMA GEMM:  C(MxN) = A(MxK) @ B(KxN) + bias
//   mode 0 : outF[row*N + col] = C                               (final GEMM)
//   mode 1 : outBF[row*ldOutBF + colOff + col] =
//              bf16( sigmoid(C) * Yf[row*N+col] )                (gate GEMM)
//   mode 2 : outBF laid out as (b,h,s,dph), value = C*scale      (projections)
// Block: 256 thr = 8 waves; tile 64x64; wave(w): rows (w&3)*16, cols (w>>2)*32.
// B tile is stored transposed in LDS so fragments load as ds_load_b128.
// ---------------------------------------------------------------------------
#define LDA  40   // 32 + 8 pad (bf16 elems), rows stay 16B aligned
#define LDBT 40   // Bt[64][40] transposed tile
__global__ __launch_bounds__(256) void gemm_bf16(
    const bf16* __restrict__ A, const bf16* __restrict__ Bm,
    const float* __restrict__ bias,
    int Mr, int Nr, int Kr, int mode, float scale,
    float* __restrict__ outF, const float* __restrict__ Yf,
    bf16* __restrict__ outBF, int ldOutBF, int colOff) {
  __shared__ bf16 As[64 * LDA];
  __shared__ bf16 Bt[64 * LDBT];
  const int tid = threadIdx.x, wave = tid >> 5, lane = tid & 31;
  const int rb = (wave & 3) * 16, cb = (wave >> 2) * 32;
  const int row0 = blockIdx.y * 64, col0 = blockIdx.x * 64;
  const int lh = (lane < 16) ? 0 : 8;     // K sub-offset per lane half (ISA layout)
  const int mloc = lane & 15;
  const int ar_ = tid >> 2, ac_ = (tid & 3) * 8;   // A staging coords
  const int br_ = tid >> 3, bc_ = (tid & 7) * 8;   // B staging coords
  v8f c0 = {}, c1 = {};

  for (int k0 = 0; k0 < Kr; k0 += 32) {
    // stage A tile 64x32 (vector copy)
    *(B8*)&As[ar_ * LDA + ac_] = *(const B8*)(A + (size_t)(row0 + ar_) * Kr + k0 + ac_);
    // stage B tile 32x64 transposed: Bt[n][k]
    {
      B8 g = *(const B8*)(Bm + (size_t)(k0 + br_) * Nr + col0 + bc_);
#pragma unroll
      for (int j = 0; j < 8; j++) Bt[(bc_ + j) * LDBT + br_] = g.h[j];
    }
    // prefetch next K tile while this one is consumed
    if (k0 + 32 < Kr) {
      __builtin_prefetch(A + (size_t)(row0 + ar_) * Kr + k0 + 32 + ac_, 0, 1);
      __builtin_prefetch(Bm + (size_t)(k0 + 32 + br_) * Nr + col0 + bc_, 0, 1);
    }
    __syncthreads();
    F16 fa, fb0, fb1;
    const bf16* arow = &As[(rb + mloc) * LDA + lh];
    fa.p.lo = *(const B8*)(arow);
    fa.p.hi = *(const B8*)(arow + 16);
    const bf16* b0row = &Bt[(cb + mloc) * LDBT + lh];
    const bf16* b1row = &Bt[(cb + 16 + mloc) * LDBT + lh];
    fb0.p.lo = *(const B8*)(b0row);  fb0.p.hi = *(const B8*)(b0row + 16);
    fb1.p.lo = *(const B8*)(b1row);  fb1.p.hi = *(const B8*)(b1row + 16);
    c0 = wmma_bf16(fa.v, fb0.v, c0);
    c1 = wmma_bf16(fa.v, fb1.v, c1);
    __syncthreads();
  }

  const int madd = (lane < 16) ? 0 : 8;
#pragma unroll
  for (int t = 0; t < 2; t++) {
    v8f cc = t ? c1 : c0;
    const int gn = col0 + cb + t * 16 + mloc;
    const float bval = bias[gn];
#pragma unroll
    for (int vv = 0; vv < 8; vv++) {
      const int gm = row0 + rb + vv + madd;
      float val = cc[vv] + bval;
      if (mode == 0) {
        outF[(size_t)gm * Nr + gn] = val;
      } else if (mode == 1) {
        float g = 1.0f / (1.0f + __expf(-val));
        float y = Yf[(size_t)gm * Nr + gn];
        outBF[(size_t)gm * ldOutBF + colOff + gn] = f2bf(g * y);
      } else {
        int bi = gm >> 9, s = gm & 511, hh = gn >> 6, dd = gn & 63;
        outBF[(((size_t)bi * Hsz + hh) * Ssz + s) * DPH + dd] = f2bf(val * scale);
      }
    }
  }
}

// ---------------------------------------------------------------------------
// hall[j][b,h,s] = v[b,h,s,:] . att_w[j,:] + att_b[j]   for j = 0..9
// ---------------------------------------------------------------------------
__global__ __launch_bounds__(256) void hall_kernel(
    const bf16* __restrict__ v, const float* __restrict__ att_w,
    const float* __restrict__ att_b, float* __restrict__ hall) {
  const int idx = blockIdx.x * 256 + threadIdx.x;   // over BHS
  const bf16* row = v + (size_t)idx * DPH;
  B8 ch[8];
#pragma unroll
  for (int j = 0; j < 8; j++) ch[j] = *(const B8*)(row + j * 8);
  float vr[64];
#pragma unroll
  for (int d = 0; d < 64; d++) vr[d] = bf2f(ch[d >> 3].h[d & 7]);
#pragma unroll
  for (int j = 0; j < 10; j++) {
    float acc = att_b[j];
#pragma unroll
    for (int d = 0; d < 64; d++) acc += vr[d] * att_w[j * 64 + d];
    hall[(size_t)j * BHS + idx] = acc;
  }
}

// ---------------------------------------------------------------------------
// Main attention: per (b, h, 16-query tile).
// scores = q.kT (WMMA) + gather(qrel) ; mask ; softmax in LDS ;
// ctx = attn@v (WMMA, vT layout) + sum_r bucket(attn,r)*rel_emb[r]
// ---------------------------------------------------------------------------
#define SPAD 4
__global__ __launch_bounds__(256) void attn_kernel(
    const bf16* __restrict__ q, const bf16* __restrict__ k,
    const bf16* __restrict__ vT, const float* __restrict__ rel_emb,
    const unsigned char* __restrict__ mask,
    float* __restrict__ Yf, bf16* __restrict__ Ybf, float* __restrict__ top) {
  __shared__ float sc[16][512 + SPAD];
  __shared__ float sbuf[16][33];
  __shared__ float qrel[16][33];
  __shared__ float qf[16][64];
  __shared__ float xch[4][16][16];

  const int b = blockIdx.z, h = blockIdx.y, q0 = blockIdx.x * 16;
  const int tid = threadIdx.x, wave = tid >> 5, lane = tid & 31;
  const int lh = (lane < 16) ? 0 : 8;
  const int madd = (lane < 16) ? 0 : 8;
  const size_t bhS = ((size_t)b * Hsz + h) * Ssz;
  const bf16* qbase = q + (bhS + q0) * DPH;

  // stage q tile as f32 for the qrel dot products
  for (int t = tid; t < 16 * 64; t += 256) qf[t >> 6][t & 63] = bf2f(qbase[t]);
  __syncthreads();

  // qrel[m][r] = q_row(m) . rel_emb[r] ; zero bucket sums
  for (int t = tid; t < 16 * 33; t += 256) {
    int m = t / 33, r = t % 33;
    sbuf[m][r] = 0.0f;
    float acc = 0.0f;
#pragma unroll
    for (int d = 0; d < 64; d++) acc += qf[m][d] * rel_emb[r * 64 + d];
    qrel[m][r] = acc;
  }

  // A fragments: this wave's q tile (K = dph, two 32-chunks)
  F16 fa0, fa1;
  {
    const bf16* arow = qbase + (size_t)(lane & 15) * DPH;
    fa0.p.lo = *(const B8*)(arow + lh);       fa0.p.hi = *(const B8*)(arow + lh + 16);
    fa1.p.lo = *(const B8*)(arow + 32 + lh);  fa1.p.hi = *(const B8*)(arow + 32 + lh + 16);
  }
  __syncthreads();   // qrel ready

  // scores: wave w owns keys [w*64, w*64+64) in 16-wide chunks
  for (int ch = 0; ch < 4; ch++) {
    const int kbase = wave * 64 + ch * 16;
    const bf16* krow = k + (bhS + kbase + (lane & 15)) * DPH; // B col = key
    F16 fb0, fb1;
    fb0.p.lo = *(const B8*)(krow + lh);       fb0.p.hi = *(const B8*)(krow + lh + 16);
    fb1.p.lo = *(const B8*)(krow + 32 + lh);  fb1.p.hi = *(const B8*)(krow + 32 + lh + 16);
    v8f c = {};
    c = wmma_bf16(fa0.v, fb0.v, c);
    c = wmma_bf16(fa1.v, fb1.v, c);
    const int key = kbase + (lane & 15);
#pragma unroll
    for (int vv = 0; vv < 8; vv++) {
      const int m = vv + madd, qg = q0 + m;
      int d = key - qg; d = d < -16 ? -16 : (d > 16 ? 16 : d);
      float s = c[vv] + qrel[m][d + 16];
      if (mask[((size_t)b * Ssz + qg) * Ssz + key]) s = -1.0e18f;
      sc[m][key] = s;
    }
  }
  __syncthreads();

  // softmax (wave w: rows 2w, 2w+1), bucket sums, top_attn
  for (int m = wave * 2; m < wave * 2 + 2; m++) {
    float mx = -3.0e38f;
    for (int kk = lane; kk < 512; kk += 32) mx = fmaxf(mx, sc[m][kk]);
#pragma unroll
    for (int o = 16; o > 0; o >>= 1) mx = fmaxf(mx, __shfl_xor(mx, o, 32));
    float sum = 0.0f;
    for (int kk = lane; kk < 512; kk += 32) {
      float p = __expf(sc[m][kk] - mx); sc[m][kk] = p; sum += p;
    }
#pragma unroll
    for (int o = 16; o > 0; o >>= 1) sum += __shfl_xor(sum, o, 32);
    const float inv = 1.0f / sum;
    const int qg = q0 + m;
    for (int kk = lane; kk < 512; kk += 32) {
      float at = sc[m][kk] * inv; sc[m][kk] = at;
      int d = kk - qg; d = d < -16 ? -16 : (d > 16 ? 16 : d);
      atomicAdd(&sbuf[m][d + 16], at);
      if (h == 0) top[((size_t)b * Ssz + qg) * Ssz + kk] = at;
    }
  }
  __syncthreads();

  // ctx = attn @ v : waves split 4 N-tiles x 2 K-halves ; v read from vT
  const int ntile = wave & 3, khalf = wave >> 2;
  const int ncol = ntile * 16 + (lane & 15);
  const bf16* vrow = vT + (((size_t)b * Hsz + h) * DPH + ncol) * Ssz;
  const int mm = lane & 15;
  v8f c = {};
  for (int ch = 0; ch < 8; ch++) {
    const int kc = khalf * 256 + ch * 32;
    F16 fa, fb;
    F4 s0 = *(const F4*)&sc[mm][kc + lh];
    F4 s1 = *(const F4*)&sc[mm][kc + lh + 4];
    F4 s2 = *(const F4*)&sc[mm][kc + lh + 16];
    F4 s3 = *(const F4*)&sc[mm][kc + lh + 20];
#pragma unroll
    for (int i = 0; i < 4; i++) {
      fa.v[i] = f2bf(s0.f[i]);     fa.v[4 + i]  = f2bf(s1.f[i]);
      fa.v[8 + i] = f2bf(s2.f[i]); fa.v[12 + i] = f2bf(s3.f[i]);
    }
    fb.p.lo = *(const B8*)(vrow + kc + lh);
    fb.p.hi = *(const B8*)(vrow + kc + lh + 16);
    c = wmma_bf16(fa.v, fb.v, c);
  }
  if (khalf == 1) {
#pragma unroll
    for (int vv = 0; vv < 8; vv++) xch[ntile][vv + madd][lane & 15] = c[vv];
  }
  __syncthreads();
  if (khalf == 0) {
#pragma unroll
    for (int vv = 0; vv < 8; vv++) c[vv] += xch[ntile][vv + madd][lane & 15];
    for (int r = 0; r < 33; r++) {               // rel contribution
      const float rl = rel_emb[r * 64 + ncol];
#pragma unroll
      for (int vv = 0; vv < 8; vv++) c[vv] += sbuf[vv + madd][r] * rl;
    }
#pragma unroll
    for (int vv = 0; vv < 8; vv++) {
      const int m = vv + madd;
      const size_t idx = ((size_t)b * Ssz + q0 + m) * Dsz + h * DPH + ncol;
      Yf[idx] = c[vv];
      Ybf[idx] = f2bf(c[vv]);
    }
  }
}

// ---------------------------------------------------------------------------
// Edge view: e = leaky_relu(h_i[q]+h_j[k]); a = softmax(where(adj, e, NEG));
// sub = a @ v  (vT layout). mode==0: adj = grh>1; mode==L: diag? g==4 : g==L
// ---------------------------------------------------------------------------
__global__ __launch_bounds__(256) void view_kernel(
    const bf16* __restrict__ vT, const float* __restrict__ hall,
    const int* __restrict__ grh, int i0, int i1, int mode,
    float* __restrict__ Yf, bf16* __restrict__ Ybf) {
  __shared__ float sc[16][512 + SPAD];
  __shared__ float hj[512];
  __shared__ float hi[16];
  __shared__ float xch[4][16][16];

  const int b = blockIdx.z, h = blockIdx.y, q0 = blockIdx.x * 16;
  const int tid = threadIdx.x, wave = tid >> 5, lane = tid & 31;
  const int lh = (lane < 16) ? 0 : 8;
  const int madd = (lane < 16) ? 0 : 8;
  const size_t bhS = ((size_t)b * Hsz + h) * Ssz;

  if (tid < 16) hi[tid] = hall[(size_t)i0 * BHS + bhS + q0 + tid];
  for (int kk = tid; kk < 512; kk += 256)
    hj[kk] = hall[(size_t)i1 * BHS + bhS + kk];
  __syncthreads();

  for (int t = tid; t < 16 * 512; t += 256) {
    const int m = t >> 9, kk = t & 511;
    float e = hi[m] + hj[kk];
    e = e > 0.0f ? e : 0.01f * e;               // leaky_relu, slope 0.01
    const int g = grh[((size_t)b * Ssz + q0 + m) * Ssz + kk];
    const bool adj = (mode == 0) ? (g > 1)
                   : ((q0 + m == kk) ? (g == 4) : (g == mode));
    sc[m][kk] = adj ? e : -9.0e15f;
  }
  __syncthreads();

  for (int m = wave * 2; m < wave * 2 + 2; m++) {
    float mx = -3.0e38f;
    for (int kk = lane; kk < 512; kk += 32) mx = fmaxf(mx, sc[m][kk]);
#pragma unroll
    for (int o = 16; o > 0; o >>= 1) mx = fmaxf(mx, __shfl_xor(mx, o, 32));
    float sum = 0.0f;
    for (int kk = lane; kk < 512; kk += 32) {
      float p = __expf(sc[m][kk] - mx); sc[m][kk] = p; sum += p;
    }
#pragma unroll
    for (int o = 16; o > 0; o >>= 1) sum += __shfl_xor(sum, o, 32);
    const float inv = 1.0f / sum;
    for (int kk = lane; kk < 512; kk += 32) sc[m][kk] *= inv;
  }
  __syncthreads();

  const int ntile = wave & 3, khalf = wave >> 2;
  const int ncol = ntile * 16 + (lane & 15);
  const bf16* vrow = vT + (((size_t)b * Hsz + h) * DPH + ncol) * Ssz;
  const int mm = lane & 15;
  v8f c = {};
  for (int ch = 0; ch < 8; ch++) {
    const int kc = khalf * 256 + ch * 32;
    F16 fa, fb;
    F4 s0 = *(const F4*)&sc[mm][kc + lh];
    F4 s1 = *(const F4*)&sc[mm][kc + lh + 4];
    F4 s2 = *(const F4*)&sc[mm][kc + lh + 16];
    F4 s3 = *(const F4*)&sc[mm][kc + lh + 20];
#pragma unroll
    for (int i = 0; i < 4; i++) {
      fa.v[i] = f2bf(s0.f[i]);     fa.v[4 + i]  = f2bf(s1.f[i]);
      fa.v[8 + i] = f2bf(s2.f[i]); fa.v[12 + i] = f2bf(s3.f[i]);
    }
    fb.p.lo = *(const B8*)(vrow + kc + lh);
    fb.p.hi = *(const B8*)(vrow + kc + lh + 16);
    c = wmma_bf16(fa.v, fb.v, c);
  }
  if (khalf == 1) {
#pragma unroll
    for (int vv = 0; vv < 8; vv++) xch[ntile][vv + madd][lane & 15] = c[vv];
  }
  __syncthreads();
  if (khalf == 0) {
#pragma unroll
    for (int vv = 0; vv < 8; vv++) {
      const float val = c[vv] + xch[ntile][vv + madd][lane & 15];
      const size_t idx = ((size_t)b * Ssz + q0 + vv + madd) * Dsz + h * DPH + ncol;
      Yf[idx] = val;
      Ybf[idx] = f2bf(val);
    }
  }
}

// ---------------------------------------------------------------------------
// Host-side orchestration
// ---------------------------------------------------------------------------
extern "C" void kernel_launch(void* const* d_in, const int* in_sizes, int n_in,
                              void* d_out, int out_size, void* d_ws, size_t ws_size,
                              hipStream_t stream) {
  (void)in_sizes; (void)n_in; (void)out_size; (void)ws_size;
  const float* f_key   = (const float*)d_in[0];
  const float* f_value = (const float*)d_in[1];
  const float* f_query = (const float*)d_in[2];
  const int*   grh     = (const int*)d_in[3];
  const unsigned char* mask = (const unsigned char*)d_in[4];
  const float* Wq = (const float*)d_in[5];  const float* bq = (const float*)d_in[6];
  const float* Wk = (const float*)d_in[7];  const float* bk = (const float*)d_in[8];
  const float* Wv = (const float*)d_in[9];  const float* bv = (const float*)d_in[10];
  const float* rel_emb = (const float*)d_in[11];
  const float* att_w   = (const float*)d_in[12];
  const float* att_b   = (const float*)d_in[13];
  const float* gate_w  = (const float*)d_in[14];
  const float* gate_b  = (const float*)d_in[15];
  const float* sub_w   = (const float*)d_in[16];
  const float* sub_b   = (const float*)d_in[17];
  float* out = (float*)d_out;
  float* top = out + (size_t)MM * Dsz;   // top_attn after (B,S,D) output

  // workspace carve-out (deterministic pointer math)
  size_t off = 0;
  char* base = (char*)d_ws;
  auto take = [&](size_t bytes) -> char* {
    char* p = base + off; off += (bytes + 255) & ~(size_t)255; return p;
  };
  bf16* xq  = (bf16*)take((size_t)MM * Dsz * 2);
  bf16* xk  = (bf16*)take((size_t)MM * Dsz * 2);
  bf16* xv  = (bf16*)take((size_t)MM * Dsz * 2);
  bf16* Wqb = (bf16*)take((size_t)Dsz * Dsz * 2);
  bf16* Wkb = (bf16*)take((size_t)Dsz * Dsz * 2);
  bf16* Wvb = (bf16*)take((size_t)Dsz * Dsz * 2);
  bf16* gwb = (bf16*)take((size_t)6 * Dsz * Dsz * 2);
  bf16* swb = (bf16*)take((size_t)CCdim * Dsz * 2);
  bf16* qb  = (bf16*)take((size_t)MM * Dsz * 2);
  bf16* kb  = (bf16*)take((size_t)MM * Dsz * 2);
  bf16* vb  = (bf16*)take((size_t)MM * Dsz * 2);
  bf16* vbT = (bf16*)take((size_t)MM * Dsz * 2);
  float* hall = (float*)take((size_t)10 * BHS * 4);
  float* Yf   = (float*)take((size_t)MM * Dsz * 4);
  bf16*  Ybf  = (bf16*)take((size_t)MM * Dsz * 2);
  bf16*  cat  = (bf16*)take((size_t)MM * CCdim * 2);

  const dim3 blk(256);
  const dim3 cvg(1024);
  // 1) f32 -> bf16 conversions
  cvt_f32_bf16<<<cvg, blk, 0, stream>>>(f_query, xq, MM * Dsz);
  cvt_f32_bf16<<<cvg, blk, 0, stream>>>(f_key,   xk, MM * Dsz);
  cvt_f32_bf16<<<cvg, blk, 0, stream>>>(f_value, xv, MM * Dsz);
  cvt_f32_bf16<<<cvg, blk, 0, stream>>>(Wq, Wqb, Dsz * Dsz);
  cvt_f32_bf16<<<cvg, blk, 0, stream>>>(Wk, Wkb, Dsz * Dsz);
  cvt_f32_bf16<<<cvg, blk, 0, stream>>>(Wv, Wvb, Dsz * Dsz);
  cvt_f32_bf16<<<cvg, blk, 0, stream>>>(gate_w, gwb, 6 * Dsz * Dsz);
  cvt_f32_bf16<<<cvg, blk, 0, stream>>>(sub_w,  swb, CCdim * Dsz);

  // 2) Q/K/V projections (mode 2: permuted bf16 store; q scaled by 1/sqrt(64))
  const dim3 gg(Dsz / 64, MM / 64);
  gemm_bf16<<<gg, blk, 0, stream>>>(xq, Wqb, bq, MM, Dsz, Dsz, 2, 0.125f,
                                    nullptr, nullptr, qb, 0, 0);
  gemm_bf16<<<gg, blk, 0, stream>>>(xk, Wkb, bk, MM, Dsz, Dsz, 2, 1.0f,
                                    nullptr, nullptr, kb, 0, 0);
  gemm_bf16<<<gg, blk, 0, stream>>>(xv, Wvb, bv, MM, Dsz, Dsz, 2, 1.0f,
                                    nullptr, nullptr, vb, 0, 0);
  transpose_v<<<dim3(Bsz * Hsz), blk, 0, stream>>>(vb, vbT);

  // 3) GAT scalar features
  hall_kernel<<<dim3(BHS / 256), blk, 0, stream>>>(vb, att_w, att_b, hall);

  // 4) main attention -> context (Yf/Ybf) + top_attn
  const dim3 ag(Ssz / 16, Hsz, Bsz);
  attn_kernel<<<ag, blk, 0, stream>>>(qb, kb, vbT, rel_emb, mask, Yf, Ybf, top);

  // 5) gate_context = sigmoid(ctx@gw0+gb0)*ctx  -> concat section 0
  gemm_bf16<<<gg, blk, 0, stream>>>(Ybf, gwb, gate_b,
                                    MM, Dsz, Dsz, 1, 1.0f, nullptr, Yf, cat, CCdim, 0);

  // 6) four edge views, each followed by its gate GEMM into concat
  struct VP { int i0, i1, mode, gi, sec; };
  const VP vps[4] = { {6, 7, 0, 5, 1},    // grh>1 view, gate_w[5]
                      {0, 1, 2, 1, 2},    // label 2,   gate_w[1]
                      {2, 3, 3, 2, 3},    // label 3,   gate_w[2]
                      {4, 5, 4, 3, 4} };  // label 4,   gate_w[3]
  for (int i = 0; i < 4; i++) {
    view_kernel<<<ag, blk, 0, stream>>>(vbT, hall, grh, vps[i].i0, vps[i].i1,
                                        vps[i].mode, Yf, Ybf);
    gemm_bf16<<<gg, blk, 0, stream>>>(Ybf, gwb + (size_t)vps[i].gi * Dsz * Dsz,
                                      gate_b + (size_t)vps[i].gi * Dsz,
                                      MM, Dsz, Dsz, 1, 1.0f,
                                      nullptr, Yf, cat, CCdim, vps[i].sec * Dsz);
  }

  // 7) final: out = concat(4096x5120) @ sub_w(5120x1024) + sub_b
  gemm_bf16<<<gg, blk, 0, stream>>>(cat, swb, sub_b, MM, Dsz, CCdim, 0, 1.0f,
                                    out, nullptr, nullptr, 0, 0);
}
